// GNN_11227044512398
// MI455X (gfx1250) — matrix operand order repaired and checked by reading, not compiled, not run on hardware
//
#include <hip/hip_runtime.h>
#include <hip/hip_bf16.h>
#include <stdint.h>
#include <math.h>

// ---------------------------------------------------------------------------
// MI455X (gfx1250) implementation of the 4-block MetaLayer GNN.
// Edge/node MLPs run as 16-row tiles through v_wmma_f32_16x16x32_f16:
//   A (16 x Kpad fp16) gathered into LDS, B (weights, transposed fp16) in LDS,
//   fragments packed per the CDNA5 ISA 16-bit A / B lane layouts (CDNA5 ISA).
// Weights are staged to LDS via async global->LDS DMA (ASYNCcnt) when the
// toolchain exposes the gfx1250 builtin; next-tile streaming operands are
// prefetched with global_prefetch_b8. Segment stats use global f32 atomics;
// BN stats use ds_add_f32 partials + one global atomic per block.
// ---------------------------------------------------------------------------

typedef __attribute__((ext_vector_type(16))) _Float16 v16h;
typedef __attribute__((ext_vector_type(8)))  float    v8f;
typedef _Float16 h16;

#define N_S   200000
#define N_T   100000
#define NEDGE 2000000
#define WAVES 4

#define AS1 __attribute__((address_space(1)))
#define AS3 __attribute__((address_space(3)))

#if __has_builtin(__builtin_amdgcn_global_load_async_to_lds_b32)
#define HAVE_ASYNC_LDS 1
#else
#define HAVE_ASYNC_LDS 0
#endif

// --- wave-local LDS ordering: CDNA5 split counters -------------------------
__device__ inline void wave_lds_sync() {
  __builtin_amdgcn_wave_barrier();
  asm volatile("s_wait_dscnt 0x0" ::: "memory");
  __builtin_amdgcn_wave_barrier();
}

// Async DMA of n contiguous f32 from global into an LDS staging buffer.
// Tracked with ASYNCcnt; each wave waits on its own counter, the caller's
// __syncthreads() makes the LDS writes visible block-wide.
__device__ inline void async_fetch(float* dstLDS, const float* srcG, int n,
                                   int tid, int nth) {
#if HAVE_ASYNC_LDS
  for (int i = tid; i < n; i += nth) {
    __builtin_amdgcn_global_load_async_to_lds_b32(
        (AS1 int*)(srcG + i), (AS3 int*)(dstLDS + i), 0, 0);
  }
  asm volatile("s_wait_asynccnt 0x0" ::: "memory");
#else
  for (int i = tid; i < n; i += nth) dstLDS[i] = srcG[i];
#endif
}

__device__ inline v8f wmma_f16f32(v16h a, v16h b, v8f c) {
  return __builtin_amdgcn_wmma_f32_16x16x32_f16(false, a, false, b,
                                                (short)0, c, false, false);
}

// A fragment: 16x32 fp16 tile, row-major LDS [16][kp] halves.
// Lane layout (ISA 7.12.2): lanes 0-15 M=lane, v0..3 K=2v,2v+1, v4..7 K=16+...
// lanes 16-31 same M, K offset +8 (low half) / +24 (high half).
__device__ inline v16h load_a_frag(const h16* A, int kp, int lane, int kb) {
  const int m = lane & 15, g = lane >> 4;
  const uint32_t* row = (const uint32_t*)(A + m * kp) + kb * 16;
  union { v16h h; uint32_t u[8]; } r;
#pragma unroll
  for (int v = 0; v < 8; ++v) {
    int k = (v < 4) ? (g * 8 + 2 * v) : (16 + g * 8 + 2 * (v - 4));
    r.u[v] = row[k >> 1];
  }
  return r.h;
}

// B fragment: 32x16 fp16, weights staged TRANSPOSED in LDS: Wt[n][kp] halves.
// Lanes 0-15 hold K=0-7,16-23 of column n=lane; lanes 16-31 hold K=8-15,24-31.
__device__ inline v16h load_b_frag(const h16* Wt, int kp, int lane, int kb) {
  const int n = lane & 15, g = lane >> 4;
  const uint32_t* col = (const uint32_t*)(Wt + n * kp) + kb * 16;
  union { v16h h; uint32_t u[8]; } r;
#pragma unroll
  for (int v = 0; v < 8; ++v) r.u[v] = col[(v < 4) ? (g * 4 + v) : (g * 4 + v + 4)];
  return r.h;
}

// Convert staged W[din][dout] (row-major f32, already in LDS) into the fp16
// transposed + zero-padded WMMA B layout, tiled by 16 output columns.
__device__ inline void stage_w(h16* dst, const float* W, int din, int dout,
                               int kp, int ntiles, int tid, int nth) {
  int total = ntiles * 16 * kp;
  for (int idx = tid; idx < total; idx += nth) {
    int ct = idx / (16 * kp), rem = idx % (16 * kp);
    int n = rem / kp, k = rem % kp;
    int col = ct * 16 + n;
    float w = (k < din && col < dout) ? W[k * dout + col] : 0.f;
    dst[idx] = (h16)w;
  }
}

// ---------------------------------------------------------------------------
// Edge-level fused gather + 2-layer MLP (leaky 0.1) + mode-specific emit.
// MODE 0: edge update (35->10->10), write edge_attr in place
// MODE 1: s1 pass A  (15->15->15), atomics: sum, sum^2, count per src
// MODE 2: s1 pass B  (15->15->15), atomics: diff^3, diff^4 per src (a0 = mean)
// MODE 3: t1         (20->20->20), atomics: sum per tgt
// MODE 4: final head (35->5->5), softmax/expected-class/noise/sigmoid -> o0
// ---------------------------------------------------------------------------
template <int MODE>
__global__ __launch_bounds__(128) void edge_kernel(
    const int* __restrict__ src, const int* __restrict__ tgt,
    const float* __restrict__ xs, const float* __restrict__ xt,
    float* __restrict__ ea, const float* __restrict__ uu,
    const float* __restrict__ W1, const float* __restrict__ B1,
    const float* __restrict__ W2, const float* __restrict__ B2,
    float* __restrict__ o0, float* __restrict__ o1, float* __restrict__ o2,
    const float* __restrict__ a0, int nE) {
  constexpr int DIN  = (MODE == 0 || MODE == 4) ? 35 : (MODE == 3 ? 20 : 15);
  constexpr int DH   = (MODE == 0) ? 10 : (MODE == 4 ? 5 : (MODE == 3 ? 20 : 15));
  constexpr int DOUT = DH;
  constexpr int KP1  = (DIN <= 32) ? 32 : 64;
  constexpr int KP2  = 32;
  constexpr int NT1  = (DH <= 16) ? 1 : 2;
  constexpr int NT2  = (DOUT <= 16) ? 1 : 2;
  constexpr int WSTG = (DIN * DH > DH * DOUT) ? DIN * DH : DH * DOUT;

  __shared__ float sStage[WSTG];
  __shared__ h16   sW1[NT1 * 16 * KP1];
  __shared__ h16   sW2[NT2 * 16 * KP2];
  __shared__ h16   sA1[WAVES][16 * KP1];
  __shared__ h16   sA2[WAVES][16 * KP2];
  __shared__ float sD[(MODE == 4) ? (WAVES * 256) : 1];

  const int tid = threadIdx.x;
  // weights: async DMA to LDS staging, then convert to WMMA B layout
  async_fetch(sStage, W1, DIN * DH, tid, blockDim.x);
  __syncthreads();
  stage_w(sW1, sStage, DIN, DH, KP1, NT1, tid, blockDim.x);
  __syncthreads();
  async_fetch(sStage, W2, DH * DOUT, tid, blockDim.x);
  __syncthreads();
  stage_w(sW2, sStage, DH, DOUT, KP2, NT2, tid, blockDim.x);
  __syncthreads();

  const int lane = tid & 31, wave = tid >> 5;
  const int g = lane >> 4, r = lane & 15, n = lane & 15;
  h16* A1 = sA1[wave];
  h16* A2 = sA2[wave];

  // zero the A2 columns never written by layer-1 restage
  if constexpr (NT1 * 16 < KP2) {
    constexpr int PADC = KP2 - NT1 * 16;
    for (int idx = lane; idx < 16 * PADC; idx += 32) {
      int row = idx / PADC, cc = NT1 * 16 + idx % PADC;
      A2[row * KP2 + cc] = (h16)0.f;
    }
  }
  wave_lds_sync();

  const int ntiles  = (nE + 15) / 16;
  const int wid     = blockIdx.x * WAVES + wave;
  const int wstride = gridDim.x * WAVES;

  for (int tile = wid; tile < ntiles; tile += wstride) {
    const int e0 = tile * 16;
    // ---- gather 16 rows into LDS (lane pair per row) ----
    {
      const int er = (e0 + r < nE) ? (e0 + r) : (nE - 1);
      const int s = src[er], t = tgt[er];
      h16* Arow = A1 + r * KP1;
      constexpr int HK = KP1 / 2;
#pragma unroll
      for (int j = 0; j < HK; ++j) {
        const int f = g * HK + j;
        float x = 0.f;
        if constexpr (MODE == 0 || MODE == 4) {
          if (f < 10)      x = xs[(size_t)s * 10 + f];
          else if (f < 15) x = xt[(size_t)t * 5 + (f - 10)];
          else if (f < 25) x = ea[(size_t)er * 10 + (f - 15)];
          else if (f < 35) x = uu[f - 25];
        } else if constexpr (MODE == 1 || MODE == 2) {
          if (f < 5)       x = xt[(size_t)t * 5 + f];
          else if (f < 15) x = ea[(size_t)er * 10 + (f - 5)];
        } else {  // MODE 3
          if (f < 10)      x = xs[(size_t)s * 10 + f];
          else if (f < 20) x = ea[(size_t)er * 10 + (f - 10)];
        }
        Arow[f] = (h16)x;
      }
      // prefetch the next grid-stride tile's streaming operands
      // (emits global_prefetch_b8; speculative)
      const int tnext = tile + wstride;
      if (tnext < ntiles) {
        const int en = (tnext * 16 + r < nE) ? (tnext * 16 + r) : (nE - 1);
        __builtin_prefetch(&ea[(size_t)en * 10], 0, 1);
        if (g) __builtin_prefetch(&src[en], 0, 1);
        else   __builtin_prefetch(&tgt[en], 0, 1);
      }
    }
    wave_lds_sync();

    // ---- layer 1 ----
    v8f acc1[NT1];
#pragma unroll
    for (int ct = 0; ct < NT1; ++ct) {
      v8f c = {};
#pragma unroll
      for (int kb = 0; kb < KP1 / 32; ++kb) {
        v16h av = load_a_frag(A1, KP1, lane, kb);
        v16h bv = load_b_frag(sW1 + ct * 16 * KP1, KP1, lane, kb);
        c = wmma_f16f32(av, bv, c);
      }
      acc1[ct] = c;
    }
    // bias + leaky relu, restage as fp16 A2
#pragma unroll
    for (int ct = 0; ct < NT1; ++ct) {
      const int col = ct * 16 + n;
      const float bias = (col < DH) ? B1[col] : 0.f;
#pragma unroll
      for (int i = 0; i < 8; ++i) {
        const int row = i + 8 * g;
        float x = acc1[ct][i] + bias;
        x = (x >= 0.f) ? x : 0.1f * x;
        A2[row * KP2 + col] = (col < DH) ? (h16)x : (h16)0.f;
      }
    }
    wave_lds_sync();

    // ---- layer 2 ----
    v8f acc2[NT2];
#pragma unroll
    for (int ct = 0; ct < NT2; ++ct) {
      v8f c = {};
      v16h av = load_a_frag(A2, KP2, lane, 0);
      v16h bv = load_b_frag(sW2 + ct * 16 * KP2, KP2, lane, 0);
      acc2[ct] = wmma_f16f32(av, bv, c);
    }

    // ---- emit ----
    if constexpr (MODE == 0) {
#pragma unroll
      for (int i = 0; i < 8; ++i) {
        const int er = e0 + i + 8 * g;
        if (er < nE && n < DOUT) ea[(size_t)er * 10 + n] = acc2[0][i] + B2[n];
      }
    } else if constexpr (MODE == 1) {
      const float bn_ = (n < DOUT) ? B2[n] : 0.f;
#pragma unroll
      for (int i = 0; i < 8; ++i) {
        const int er = e0 + i + 8 * g;
        if (er < nE) {
          const int sr = src[er];
          if (n < DOUT) {
            float v = acc2[0][i] + bn_;
            atomicAdd(&o0[(size_t)sr * 15 + n], v);
            atomicAdd(&o1[(size_t)sr * 15 + n], v * v);
          }
          if (n == 0) atomicAdd(&o2[sr], 1.f);
        }
      }
    } else if constexpr (MODE == 2) {
      const float bn_ = (n < DOUT) ? B2[n] : 0.f;
#pragma unroll
      for (int i = 0; i < 8; ++i) {
        const int er = e0 + i + 8 * g;
        if (er < nE && n < DOUT) {
          const int sr = src[er];
          float v = acc2[0][i] + bn_;
          float dd = v - a0[(size_t)sr * 15 + n];
          float d3 = dd * dd * dd;
          atomicAdd(&o0[(size_t)sr * 15 + n], d3);
          atomicAdd(&o1[(size_t)sr * 15 + n], d3 * dd);
        }
      }
    } else if constexpr (MODE == 3) {
#pragma unroll
      for (int ct = 0; ct < NT2; ++ct) {
        const int col = ct * 16 + n;
        const float bn_ = (col < DOUT) ? B2[col] : 0.f;
#pragma unroll
        for (int i = 0; i < 8; ++i) {
          const int er = e0 + i + 8 * g;
          if (er < nE && col < DOUT) {
            const int tr = tgt[er];
            atomicAdd(&o0[(size_t)tr * 20 + col], acc2[ct][i] + bn_);
          }
        }
      }
    } else {  // MODE 4
      float* D = sD + wave * 256;
#pragma unroll
      for (int i = 0; i < 8; ++i) {
        const int row = i + 8 * g;
        if (n < 5) D[row * 16 + n] = acc2[0][i] + B2[n];
      }
      wave_lds_sync();
      if (lane < 16) {
        const int er = e0 + lane;
        if (er < nE) {
          float v0 = D[lane * 16 + 0], v1 = D[lane * 16 + 1];
          float v2 = D[lane * 16 + 2], v3 = D[lane * 16 + 3];
          float v4 = D[lane * 16 + 4];
          float mx = fmaxf(fmaxf(fmaxf(v0, v1), fmaxf(v2, v3)), v4);
          float p0 = __expf(v0 - mx), p1 = __expf(v1 - mx), p2 = __expf(v2 - mx);
          float p3 = __expf(v3 - mx), p4 = __expf(v4 - mx);
          float s = p0 + p1 + p2 + p3 + p4;
          float time = (p1 + 2.f * p2 + 3.f * p3 + 4.f * p4) / s;
          time += 0.3f * (a0[er] - 0.5f);
          float it = floorf(time);
          float z = 20.f * (time - 0.5f - it);
          o0[er] = it + 1.f / (1.f + __expf(-z));
        }
      }
      wave_lds_sync();
    }
  }
}

// ---------------------------------------------------------------------------
// Node-level fused gather + MLP.
// MODE 0: s2 (81->10->10) on N_S, h = [x_s, cnt, a, b, c, d, u] -> x_s in place
// MODE 1: t2 (35->5->5)   on N_T, h = [x_t, tsum, u]            -> x_t in place
// ---------------------------------------------------------------------------
template <int MODE>
__global__ __launch_bounds__(128) void node_kernel(
    float* __restrict__ x, const float* __restrict__ p0,
    const float* __restrict__ p1, const float* __restrict__ p2,
    const float* __restrict__ p3, const float* __restrict__ p4,
    const float* __restrict__ uu,
    const float* __restrict__ W1, const float* __restrict__ B1,
    const float* __restrict__ W2, const float* __restrict__ B2, int nV) {
  constexpr int DIN  = (MODE == 0) ? 81 : 35;
  constexpr int DX   = (MODE == 0) ? 10 : 5;
  constexpr int DH   = (MODE == 0) ? 10 : 5;
  constexpr int DOUT = DH;
  constexpr int KP1  = (MODE == 0) ? 96 : 64;
  constexpr int KP2  = 32;
  constexpr int WSTG = DIN * DH;

  __shared__ float sStage[WSTG];
  __shared__ h16 sW1[16 * KP1];
  __shared__ h16 sW2[16 * KP2];
  __shared__ h16 sA1[WAVES][16 * KP1];
  __shared__ h16 sA2[WAVES][16 * KP2];

  const int tid = threadIdx.x;
  async_fetch(sStage, W1, DIN * DH, tid, blockDim.x);
  __syncthreads();
  stage_w(sW1, sStage, DIN, DH, KP1, 1, tid, blockDim.x);
  __syncthreads();
  async_fetch(sStage, W2, DH * DOUT, tid, blockDim.x);
  __syncthreads();
  stage_w(sW2, sStage, DH, DOUT, KP2, 1, tid, blockDim.x);
  __syncthreads();

  const int lane = tid & 31, wave = tid >> 5;
  const int g = lane >> 4, r = lane & 15, n = lane & 15;
  h16* A1 = sA1[wave];
  h16* A2 = sA2[wave];

  for (int idx = lane; idx < 16 * 16; idx += 32) {  // zero A2 cols 16..31
    int row = idx / 16, cc = 16 + idx % 16;
    A2[row * KP2 + cc] = (h16)0.f;
  }
  wave_lds_sync();

  const int ntiles  = (nV + 15) / 16;
  const int wid     = blockIdx.x * WAVES + wave;
  const int wstride = gridDim.x * WAVES;

  for (int tile = wid; tile < ntiles; tile += wstride) {
    const int v0 = tile * 16;
    {
      const int vr = (v0 + r < nV) ? (v0 + r) : (nV - 1);
      h16* Arow = A1 + r * KP1;
      constexpr int HK = KP1 / 2;
      float cntv = 0.f, m = 1.f;
      if constexpr (MODE == 0) { cntv = p0[vr]; m = fmaxf(cntv, 1.f); }
#pragma unroll
      for (int j = 0; j < HK; ++j) {
        const int f = g * HK + j;
        float xv = 0.f;
        if constexpr (MODE == 0) {
          if (f < 10)       xv = x[(size_t)vr * 10 + f];
          else if (f == 10) xv = cntv;
          else if (f < 26)  xv = p1[(size_t)vr * 15 + (f - 11)];
          else if (f < 41)  xv = p2[(size_t)vr * 15 + (f - 26)];
          else if (f < 56) {
            float b = p2[(size_t)vr * 15 + (f - 41)];
            xv = (p3[(size_t)vr * 15 + (f - 41)] / m) / (b * b * b);
          } else if (f < 71) {
            float b = p2[(size_t)vr * 15 + (f - 56)];
            float b2 = b * b;
            xv = (p4[(size_t)vr * 15 + (f - 56)] / m) / (b2 * b2);
          } else if (f < 81) xv = uu[f - 71];
        } else {
          if (f < 5)       xv = x[(size_t)vr * 5 + f];
          else if (f < 25) xv = p0[(size_t)vr * 20 + (f - 5)];
          else if (f < 35) xv = uu[f - 25];
        }
        Arow[f] = (h16)xv;
      }
      const int tnext = tile + wstride;
      if (tnext < ntiles) {
        const int vn = (tnext * 16 + r < nV) ? (tnext * 16 + r) : (nV - 1);
        __builtin_prefetch(&x[(size_t)vn * DX], 0, 1);
        if constexpr (MODE == 0) __builtin_prefetch(&p1[(size_t)vn * 15], 0, 1);
        else                     __builtin_prefetch(&p0[(size_t)vn * 20], 0, 1);
      }
    }
    wave_lds_sync();

    v8f c1 = {};
#pragma unroll
    for (int kb = 0; kb < KP1 / 32; ++kb) {
      v16h av = load_a_frag(A1, KP1, lane, kb);
      v16h bv = load_b_frag(sW1, KP1, lane, kb);
      c1 = wmma_f16f32(av, bv, c1);
    }
    {
      const float bias = (n < DH) ? B1[n] : 0.f;
#pragma unroll
      for (int i = 0; i < 8; ++i) {
        const int row = i + 8 * g;
        float xv = c1[i] + bias;
        xv = (xv >= 0.f) ? xv : 0.1f * xv;
        A2[row * KP2 + n] = (n < DH) ? (h16)xv : (h16)0.f;
      }
    }
    wave_lds_sync();

    v8f c2 = {};
    {
      v16h av = load_a_frag(A2, KP2, lane, 0);
      v16h bv = load_b_frag(sW2, KP2, lane, 0);
      c2 = wmma_f16f32(av, bv, c2);
    }
#pragma unroll
    for (int i = 0; i < 8; ++i) {
      const int vr = v0 + i + 8 * g;
      if (vr < nV && n < DOUT) x[(size_t)vr * DX + n] = c2[i] + B2[n];
    }
  }
}

// ---- s-model stat finalization: sum->mean, sumsq->std ---------------------
__global__ void finalize_s_kernel(float* __restrict__ sum,
                                  float* __restrict__ sumsq,
                                  const float* __restrict__ cnt, int nNode) {
  int i = blockIdx.x * blockDim.x + threadIdx.x;
  if (i < nNode * 15) {
    float m = fmaxf(cnt[i / 15], 1.f);
    float a = sum[i] / m;
    float m2 = sumsq[i] / m;
    sum[i] = a;
    sumsq[i] = sqrtf(1e-6f + fmaxf(m2 - a * a, 0.f));
  }
}

// ---- column sum / sumsq reduction (LDS ds_add_f32 + one global atomic) ----
template <int C>
__global__ __launch_bounds__(256) void colstats_kernel(
    const float* __restrict__ x, int n, float* __restrict__ red) {
  __shared__ float ssum[16], ssq[16];
  if (threadIdx.x < 16) { ssum[threadIdx.x] = 0.f; ssq[threadIdx.x] = 0.f; }
  __syncthreads();
  float ls[C], lq[C];
#pragma unroll
  for (int c = 0; c < C; ++c) { ls[c] = 0.f; lq[c] = 0.f; }
  for (int i = blockIdx.x * blockDim.x + threadIdx.x; i < n;
       i += gridDim.x * blockDim.x) {
#pragma unroll
    for (int c = 0; c < C; ++c) {
      float v = x[(size_t)i * C + c];
      ls[c] += v;
      lq[c] += v * v;
    }
  }
#pragma unroll
  for (int c = 0; c < C; ++c) {
    atomicAdd(&ssum[c], ls[c]);
    atomicAdd(&ssq[c], lq[c]);
  }
  __syncthreads();
  if (threadIdx.x < C) {
    atomicAdd(&red[threadIdx.x], ssum[threadIdx.x]);
    atomicAdd(&red[16 + threadIdx.x], ssq[threadIdx.x]);
  }
}

// ---- BatchNorm apply (train-mode, biased var) -----------------------------
template <int C>
__global__ void bn_apply_kernel(float* __restrict__ x, int n,
                                const float* __restrict__ red,
                                const float* __restrict__ gg,
                                const float* __restrict__ bb) {
  int i = blockIdx.x * blockDim.x + threadIdx.x;
  if (i < n * C) {
    int c = i % C;
    float m = red[c] / (float)n;
    float v = red[16 + c] / (float)n - m * m;
    x[i] = (x[i] - m) * rsqrtf(v + 1e-5f) * gg[c] + bb[c];
  }
}

// ---- global model: u' = MLP25->10->10([u, mean(x_s), mean(x_t)]), B=1 -----
__global__ void g_kernel(float* __restrict__ uu, const float* __restrict__ redS,
                         const float* __restrict__ redT,
                         const float* __restrict__ W1, const float* __restrict__ B1,
                         const float* __restrict__ W2, const float* __restrict__ B2) {
  __shared__ float xin[25], h[10];
  int t = threadIdx.x;
  if (t < 10)                xin[t] = uu[t];
  else if (t < 20)           xin[t] = redS[t - 10] / (float)N_S;
  else if (t < 25)           xin[t] = redT[t - 20] / (float)N_T;
  __syncthreads();
  if (t < 10) {
    float s = B1[t];
    for (int k = 0; k < 25; ++k) s += xin[k] * W1[k * 10 + t];
    h[t] = (s >= 0.f) ? s : 0.1f * s;
  }
  __syncthreads();
  if (t < 10) {
    float s = B2[t];
    for (int k = 0; k < 10; ++k) s += h[k] * W2[k * 10 + t];
    uu[t] = s;
  }
}

// ---------------------------------------------------------------------------
extern "C" void kernel_launch(void* const* d_in, const int* in_sizes, int n_in,
                              void* d_out, int out_size, void* d_ws,
                              size_t ws_size, hipStream_t stream) {
  (void)in_sizes; (void)n_in; (void)out_size; (void)ws_size;
  const float* in_xs = (const float*)d_in[0];
  const float* in_xt = (const float*)d_in[1];
  const int*   src   = (const int*)d_in[2];
  const int*   tgt   = (const int*)d_in[3];
  const float* in_ea = (const float*)d_in[4];
  const float* in_u  = (const float*)d_in[5];
  const float* noise = (const float*)d_in[9];
  auto P = [&](int i) { return (const float*)d_in[i]; };

  // workspace carve
  size_t off = 0;
  auto carve = [&](size_t bytes) -> float* {
    float* p = (float*)((char*)d_ws + off);
    off = (off + bytes + 255) & ~(size_t)255;
    return p;
  };
  float* xs   = carve((size_t)N_S * 10 * 4);
  float* xt   = carve((size_t)N_T * 5 * 4);
  float* ea   = carve((size_t)NEDGE * 10 * 4);
  float* uu   = carve(64);
  float* ssum = carve((size_t)N_S * 15 * 4);
  float* ssq  = carve((size_t)N_S * 15 * 4);
  float* scnt = carve((size_t)N_S * 4);
  float* c3   = carve((size_t)N_S * 15 * 4);
  float* c4   = carve((size_t)N_S * 15 * 4);
  float* tsum = carve((size_t)N_T * 20 * 4);
  float* red  = carve(64 * 4);

  (void)hipMemcpyAsync(xs, in_xs, (size_t)N_S * 10 * 4, hipMemcpyDeviceToDevice, stream);
  (void)hipMemcpyAsync(xt, in_xt, (size_t)N_T * 5 * 4, hipMemcpyDeviceToDevice, stream);
  (void)hipMemcpyAsync(ea, in_ea, (size_t)NEDGE * 10 * 4, hipMemcpyDeviceToDevice, stream);
  (void)hipMemcpyAsync(uu, in_u, 10 * 4, hipMemcpyDeviceToDevice, stream);

  const dim3 tb(128);
  const int egrid = 1024, sgrid = 512, tgrid = 256;

  for (int b = 0; b < 4; ++b) {
    // params: tree-flatten alphabetical: blocks[b] keys edge,g,s1,s2,t1,t2;
    // each MLP W1,W2,b1,b2
    int blk = 10 + 24 * b;
    const float *eW1 = P(blk + 0),  *eW2 = P(blk + 1),  *eB1 = P(blk + 2),  *eB2 = P(blk + 3);
    const float *gW1 = P(blk + 4),  *gW2 = P(blk + 5),  *gB1 = P(blk + 6),  *gB2 = P(blk + 7);
    const float *s1W1 = P(blk + 8), *s1W2 = P(blk + 9), *s1B1 = P(blk + 10), *s1B2 = P(blk + 11);
    const float *s2W1 = P(blk + 12),*s2W2 = P(blk + 13),*s2B1 = P(blk + 14), *s2B2 = P(blk + 15);
    const float *t1W1 = P(blk + 16),*t1W2 = P(blk + 17),*t1B1 = P(blk + 18), *t1B2 = P(blk + 19);
    const float *t2W1 = P(blk + 20),*t2W2 = P(blk + 21),*t2B1 = P(blk + 22), *t2B2 = P(blk + 23);
    int bnb = 10 + 96 + 6 * b;  // bn[b] keys e,xs,xt; each b,g
    const float *bnEb = P(bnb + 0), *bnEg = P(bnb + 1);
    const float *bnSb = P(bnb + 2), *bnSg = P(bnb + 3);
    const float *bnTb = P(bnb + 4), *bnTg = P(bnb + 5);

    // edge update (in place)
    edge_kernel<0><<<egrid, tb, 0, stream>>>(src, tgt, xs, xt, ea, uu,
        eW1, eB1, eW2, eB2, nullptr, nullptr, nullptr, nullptr, NEDGE);

    // s-model: pass A (sum, sumsq, count)
    (void)hipMemsetAsync(ssum, 0, (size_t)N_S * 15 * 4, stream);
    (void)hipMemsetAsync(ssq, 0, (size_t)N_S * 15 * 4, stream);
    (void)hipMemsetAsync(scnt, 0, (size_t)N_S * 4, stream);
    edge_kernel<1><<<egrid, tb, 0, stream>>>(src, tgt, xs, xt, ea, uu,
        s1W1, s1B1, s1W2, s1B2, ssum, ssq, scnt, nullptr, NEDGE);
    finalize_s_kernel<<<(N_S * 15 + 255) / 256, 256, 0, stream>>>(ssum, ssq, scnt, N_S);
    // pass B (central 3rd/4th moments)
    (void)hipMemsetAsync(c3, 0, (size_t)N_S * 15 * 4, stream);
    (void)hipMemsetAsync(c4, 0, (size_t)N_S * 15 * 4, stream);
    edge_kernel<2><<<egrid, tb, 0, stream>>>(src, tgt, xs, xt, ea, uu,
        s1W1, s1B1, s1W2, s1B2, c3, c4, nullptr, ssum, NEDGE);
    node_kernel<0><<<sgrid, tb, 0, stream>>>(xs, scnt, ssum, ssq, c3, c4, uu,
        s2W1, s2B1, s2W2, s2B2, N_S);

    // t-model
    (void)hipMemsetAsync(tsum, 0, (size_t)N_T * 20 * 4, stream);
    edge_kernel<3><<<egrid, tb, 0, stream>>>(src, tgt, xs, xt, ea, uu,
        t1W1, t1B1, t1W2, t1B2, tsum, nullptr, nullptr, nullptr, NEDGE);
    node_kernel<1><<<tgrid, tb, 0, stream>>>(xt, tsum, nullptr, nullptr, nullptr,
        nullptr, uu, t2W1, t2B1, t2W2, t2B2, N_T);

    // g-model (node means, then 1-row MLP)
    (void)hipMemsetAsync(red, 0, 64 * 4, stream);
    colstats_kernel<10><<<512, 256, 0, stream>>>(xs, N_S, red);
    colstats_kernel<5><<<512, 256, 0, stream>>>(xt, N_T, red + 32);
    g_kernel<<<1, 32, 0, stream>>>(uu, red, red + 32, gW1, gB1, gW2, gB2);

    // BatchNorm (train mode) on x_s, x_t, edge_attr
    (void)hipMemsetAsync(red, 0, 64 * 4, stream);
    colstats_kernel<10><<<512, 256, 0, stream>>>(xs, N_S, red);
    bn_apply_kernel<10><<<(N_S * 10 + 255) / 256, 256, 0, stream>>>(xs, N_S, red, bnSg, bnSb);
    (void)hipMemsetAsync(red, 0, 64 * 4, stream);
    colstats_kernel<5><<<512, 256, 0, stream>>>(xt, N_T, red);
    bn_apply_kernel<5><<<(N_T * 5 + 255) / 256, 256, 0, stream>>>(xt, N_T, red, bnTg, bnTb);
    (void)hipMemsetAsync(red, 0, 64 * 4, stream);
    colstats_kernel<10><<<1024, 256, 0, stream>>>(ea, NEDGE, red);
    bn_apply_kernel<10><<<(NEDGE * 10 + 255) / 256, 256, 0, stream>>>(ea, NEDGE, red, bnEg, bnEb);
  }

  // final edge head + softmax expected-class + noise + sigmoid rounding
  edge_kernel<4><<<egrid, tb, 0, stream>>>(src, tgt, xs, xt, ea, uu,
      P(130), P(132), P(131), P(133), (float*)d_out, nullptr, nullptr, noise, NEDGE);
}
